// EAGLE_V2_38972533244372
// MI455X (gfx1250) — matrix operand order repaired and checked by reading, not compiled, not run on hardware
//
#include <hip/hip_runtime.h>
#include <hip/hip_bf16.h>
#include <math.h>

// ---------------- problem constants ----------------
constexpr int NB = 128;           // batch
constexpr int NN = 256;           // nodes
constexpr int NH = 768;           // bert dim
constexpr int NG = 300;           // gcn dim
constexpr int NGP = 320;          // padded gcn dim (5 * 64)
constexpr int NK = 10;            // topk
constexpr int NM = NB * NN;       // 32768 flat rows
constexpr float NEGV = -9.0e15f;
constexpr float LNEPS = 1e-5f;

typedef __bf16 bf16_t;
typedef __attribute__((ext_vector_type(16))) __bf16 v16bf;
typedef __attribute__((ext_vector_type(8)))  __bf16 v8bf;
typedef __attribute__((ext_vector_type(8)))  float  v8f;

// ---------------- wave32 helpers ----------------
__device__ inline float wave_sum(float v) {
#pragma unroll
  for (int m = 16; m >= 1; m >>= 1) v += __shfl_xor(v, m, 32);
  return v;
}
__device__ inline float wave_max(float v) {
#pragma unroll
  for (int m = 16; m >= 1; m >>= 1) v = fmaxf(v, __shfl_xor(v, m, 32));
  return v;
}
__device__ inline void wave_argmax(float &v, int &i) {
#pragma unroll
  for (int m = 16; m >= 1; m >>= 1) {
    float ov = __shfl_xor(v, m, 32);
    int   oi = __shfl_xor(i, m, 32);
    if (ov > v || (ov == v && oi < i)) { v = ov; i = oi; }
  }
}

// ---------------- WMMA fragment loaders (CDNA5 16-bit layouts) ----------------
// A (16x32 bf16): lane<16 -> M=lane, K = k0+0..7 and k0+16..23 ; lane>=16 same M, K shifted by 8
__device__ inline v16bf load_a_frag(const bf16_t* row, int k0, int half) {
  const bf16_t* p = row + k0 + half * 8;
  v8bf lo = *(const v8bf*)(p);
  v8bf hi = *(const v8bf*)(p + 16);
  v16bf f;
#pragma unroll
  for (int e = 0; e < 8; ++e) { f[e] = lo[e]; f[e + 8] = hi[e]; }
  return f;
}
// B (32x16 bf16): lane<16 -> N=lane, K = k0+0..7 and k0+16..23 ; lane>=16 same N, K shifted by 8
__device__ inline v16bf load_b_frag(const bf16_t* col, int k0, int half) {
  const bf16_t* p = col + k0 + half * 8;
  v8bf lo = *(const v8bf*)(p);
  v8bf hi = *(const v8bf*)(p + 16);
  v16bf f;
#pragma unroll
  for (int e = 0; e < 8; ++e) { f[e] = lo[e]; f[e + 8] = hi[e]; }
  return f;
}
__device__ inline v8f wmma_bf16(v16bf a, v16bf b, v8f c) {
  return __builtin_amdgcn_wmma_f32_16x16x32_bf16(false, a, false, b, (short)0, c, false, false);
}

// ---------------- prep: row norms + bf16 conversion ----------------
__global__ __launch_bounds__(256)
void k_prep_h(const float* __restrict__ h, bf16_t* __restrict__ hbf, bf16_t* __restrict__ hnbf) {
  __shared__ float red[8];
  const int row = blockIdx.x;
  const int t = threadIdx.x;
  const float* src = h + (size_t)row * NH;
  float v0 = src[t], v1 = src[t + 256], v2 = src[t + 512];
  float ss = v0 * v0 + v1 * v1 + v2 * v2;
  ss = wave_sum(ss);
  if ((t & 31) == 0) red[t >> 5] = ss;
  __syncthreads();
  float tot = 0.f;
#pragma unroll
  for (int i = 0; i < 8; ++i) tot += red[i];
  float inv = 1.0f / fmaxf(sqrtf(tot), 1e-12f);
  bf16_t* db = hbf + (size_t)row * NH;
  bf16_t* dn = hnbf + (size_t)row * NH;
  db[t] = (bf16_t)v0; db[t + 256] = (bf16_t)v1; db[t + 512] = (bf16_t)v2;
  dn[t] = (bf16_t)(v0 * inv); dn[t + 256] = (bf16_t)(v1 * inv); dn[t + 512] = (bf16_t)(v2 * inv);
}

__global__ void k_zero(float* __restrict__ p, size_t n) {
  size_t i = (size_t)blockIdx.x * blockDim.x + threadIdx.x;
  if (i < n) p[i] = 0.f;
}

// ---------------- sim = hn @ hn^T  (batched WMMA GEMM, 16x64 per wave, pipelined) ----------------
__global__ __launch_bounds__(256)
void k_sim(const bf16_t* __restrict__ hn, float* __restrict__ sim) {
  const int lane = threadIdx.x & 31, wave = threadIdx.x >> 5;
  const int half = lane >> 4, l15 = lane & 15;
  const int b = blockIdx.y;
  const int tile = blockIdx.x * 8 + wave;       // 64 wave-tiles (16 M x 4 N64)
  const int tm = tile >> 2, tnn = tile & 3;
  const bf16_t* arow  = hn + ((size_t)b * NN + tm * 16 + l15) * NH;
  const bf16_t* brow0 = hn + ((size_t)b * NN + tnn * 64 +  0 + l15) * NH;
  const bf16_t* brow1 = hn + ((size_t)b * NN + tnn * 64 + 16 + l15) * NH;
  const bf16_t* brow2 = hn + ((size_t)b * NN + tnn * 64 + 32 + l15) * NH;
  const bf16_t* brow3 = hn + ((size_t)b * NN + tnn * 64 + 48 + l15) * NH;
  v8f acc0 = {}, acc1 = {}, acc2 = {}, acc3 = {};
  v16bf a  = load_a_frag(arow, 0, half);
  v16bf b0 = load_b_frag(brow0, 0, half);
  v16bf b1 = load_b_frag(brow1, 0, half);
  v16bf b2 = load_b_frag(brow2, 0, half);
  v16bf b3 = load_b_frag(brow3, 0, half);
  constexpr int KS = NH / 32;
#pragma unroll 2
  for (int ks = 0; ks < KS; ++ks) {
    const int k0n = (ks + 1 < KS) ? (ks + 1) * 32 : 0;   // wrap: in-bounds, dead result
    v16bf an  = load_a_frag(arow, k0n, half);
    v16bf b0n = load_b_frag(brow0, k0n, half);
    v16bf b1n = load_b_frag(brow1, k0n, half);
    v16bf b2n = load_b_frag(brow2, k0n, half);
    v16bf b3n = load_b_frag(brow3, k0n, half);
    __builtin_prefetch(arow + k0n + 128, 0, 1);
    acc0 = wmma_bf16(a, b0, acc0);
    acc1 = wmma_bf16(a, b1, acc1);
    acc2 = wmma_bf16(a, b2, acc2);
    acc3 = wmma_bf16(a, b3, acc3);
    a = an; b0 = b0n; b1 = b1n; b2 = b2n; b3 = b3n;
  }
#pragma unroll
  for (int r = 0; r < 8; ++r) {
    int row = tm * 16 + r + half * 8;
    float* srow = sim + ((size_t)b * NN + row) * NN + tnn * 64 + l15;
    srow[0]  = acc0[r];
    srow[16] = acc1[r];
    srow[32] = acc2[r];
    srow[48] = acc3[r];
  }
}

// ---------------- top-k + symmetric scatter ----------------
__global__ __launch_bounds__(256)
void k_topk(const float* __restrict__ sim, float* __restrict__ adj) {
  const int lane = threadIdx.x & 31, wave = threadIdx.x >> 5;
  const int row = blockIdx.x * 8 + wave;        // 0..NM-1
  const int b = row >> 8, r = row & 255;
  const float* srow = sim + (size_t)row * NN;
  float vals[8]; int cols[8];
#pragma unroll
  for (int j = 0; j < 8; ++j) { int c = lane + 32 * j; vals[j] = srow[c]; cols[j] = c; }
  for (int t = 0; t < NK; ++t) {
    float bv = -3.0e38f; int bi = 0x7fffffff;
#pragma unroll
    for (int j = 0; j < 8; ++j)
      if (vals[j] > bv || (vals[j] == bv && cols[j] < bi)) { bv = vals[j]; bi = cols[j]; }
    wave_argmax(bv, bi);
    if ((bi & 31) == lane) vals[bi >> 5] = -3.0e38f;   // remove winner
    if (lane == 0) {
      float hv = 0.5f * bv;  // (adj + adj^T)/2 via two half-adds
      atomicAdd(&adj[((size_t)b * NN + r)  * NN + bi], hv);
      atomicAdd(&adj[((size_t)b * NN + bi) * NN + r],  hv);
    }
  }
}

// ---------------- +eye, row-normalize ----------------
__global__ __launch_bounds__(256)
void k_rownorm(float* __restrict__ adj) {
  const int lane = threadIdx.x & 31, wave = threadIdx.x >> 5;
  const int row = blockIdx.x * 8 + wave;
  const int r = row & 255;
  float* arow = adj + (size_t)row * NN;
  float vals[8]; float s = 0.f;
#pragma unroll
  for (int j = 0; j < 8; ++j) {
    int c = lane + 32 * j;
    float v = arow[c];
    if (c == r) v += 1.0f;
    vals[j] = v; s += v;
  }
  s = wave_sum(s);
  float inv = 1.0f / (s + 1e-8f);
#pragma unroll
  for (int j = 0; j < 8; ++j) arow[lane + 32 * j] = vals[j] * inv;
}

// ---------------- weight transpose + pad to bf16 ----------------
__global__ void k_tpad(const float* __restrict__ W, bf16_t* __restrict__ Wt,
                       int Ksrc, int Gsrc, int Kp) {
  size_t i = (size_t)blockIdx.x * blockDim.x + threadIdx.x;
  if (i >= (size_t)NGP * Kp) return;
  int g = (int)(i / Kp), k = (int)(i % Kp);
  float v = (g < Gsrc && k < Ksrc) ? W[(size_t)k * Gsrc + g] : 0.0f;
  Wt[i] = (bf16_t)v;
}
// fusion weight [600,300] -> Wt[320][640] with 300->320 segment padding
__global__ void k_tpad_fus(const float* __restrict__ W, bf16_t* __restrict__ Wt) {
  size_t i = (size_t)blockIdx.x * blockDim.x + threadIdx.x;
  if (i >= (size_t)NGP * 640) return;
  int g = (int)(i / 640), k = (int)(i % 640);
  int seg = k / NGP, kk = k % NGP;
  float v = (g < NG && kk < NG) ? W[(size_t)(seg * NG + kk) * NG + g] : 0.0f;
  Wt[i] = (bf16_t)v;
}

// ---------------- generic flat-M WMMA GEMM, 16x64 per wave, pipelined ----------------
// C[m,g] = sum_k A[m,k]*Bt[g,k]  (+pos_tab[positions[m],g]) (+bias[g]) (relu)
// optional transposed bf16 output: whT[b][g][j] (b=m/256, j=m%256)
// grid: x = NM/16/8 = 256 (8 waves -> 8 M-tiles), y = N64-supertiles (NGP/64 = 5)
__global__ __launch_bounds__(256)
void k_gemm(const bf16_t* __restrict__ A, int lda,
            const bf16_t* __restrict__ Bt, int ldb,
            float* __restrict__ C, int ldc, int ccols,
            int ksteps,
            const int* __restrict__ positions, const float* __restrict__ pos_tab,
            const float* __restrict__ bias, int relu,
            bf16_t* __restrict__ whT) {
  const int lane = threadIdx.x & 31, wave = threadIdx.x >> 5;
  const int half = lane >> 4, l15 = lane & 15;
  const int mt = blockIdx.x * 8 + wave;         // 2048 M-tiles
  const int g0 = blockIdx.y * 64;               // N supertile base
  const bf16_t* arow  = A  + (size_t)(mt * 16 + l15) * lda;
  const bf16_t* brow0 = Bt + (size_t)(g0 +  0 + l15) * ldb;
  const bf16_t* brow1 = Bt + (size_t)(g0 + 16 + l15) * ldb;
  const bf16_t* brow2 = Bt + (size_t)(g0 + 32 + l15) * ldb;
  const bf16_t* brow3 = Bt + (size_t)(g0 + 48 + l15) * ldb;
  v8f acc0 = {}, acc1 = {}, acc2 = {}, acc3 = {};
  v16bf a  = load_a_frag(arow, 0, half);
  v16bf b0 = load_b_frag(brow0, 0, half);
  v16bf b1 = load_b_frag(brow1, 0, half);
  v16bf b2 = load_b_frag(brow2, 0, half);
  v16bf b3 = load_b_frag(brow3, 0, half);
#pragma unroll 2
  for (int ks = 0; ks < ksteps; ++ks) {
    const int k0n = (ks + 1 < ksteps) ? (ks + 1) * 32 : 0;  // wrap: in-bounds, dead result
    v16bf an  = load_a_frag(arow, k0n, half);
    v16bf b0n = load_b_frag(brow0, k0n, half);
    v16bf b1n = load_b_frag(brow1, k0n, half);
    v16bf b2n = load_b_frag(brow2, k0n, half);
    v16bf b3n = load_b_frag(brow3, k0n, half);
    __builtin_prefetch(arow + k0n + 128, 0, 1);
    acc0 = wmma_bf16(a, b0, acc0);
    acc1 = wmma_bf16(a, b1, acc1);
    acc2 = wmma_bf16(a, b2, acc2);
    acc3 = wmma_bf16(a, b3, acc3);
    a = an; b0 = b0n; b1 = b1n; b2 = b2n; b3 = b3n;
  }
  v8f* accs[4] = { &acc0, &acc1, &acc2, &acc3 };
#pragma unroll
  for (int j = 0; j < 4; ++j) {
    const int g = g0 + j * 16 + l15;
    v8f acc = *accs[j];
#pragma unroll
    for (int r = 0; r < 8; ++r) {
      int m = mt * 16 + r + half * 8;
      float c = acc[r];
      if (pos_tab != nullptr && g < NG) c += pos_tab[(size_t)positions[m] * NG + g];
      if (bias != nullptr && g < NG) c += bias[g];
      if (relu) c = fmaxf(c, 0.0f);
      if (g < ccols) C[(size_t)m * ldc + g] = c;
      if (whT != nullptr)
        whT[(((size_t)(m >> 8)) * NGP + g) * NN + (m & 255)] = (bf16_t)c;
    }
  }
}

// ---------------- fs/fd = Wh @ asrc / adst ----------------
__global__ __launch_bounds__(256)
void k_fsfd(const float* __restrict__ Wh, const float* __restrict__ asrc,
            const float* __restrict__ adst, float* __restrict__ fs, float* __restrict__ fd) {
  const int lane = threadIdx.x & 31, wave = threadIdx.x >> 5;
  const int row = blockIdx.x * 8 + wave;
  const float* wrow = Wh + (size_t)row * NGP;
  float s1 = 0.f, s2 = 0.f;
  for (int g = lane; g < NG; g += 32) {
    float w = wrow[g];
    s1 += w * asrc[g]; s2 += w * adst[g];
  }
  s1 = wave_sum(s1); s2 = wave_sum(s2);
  if (lane == 0) { fs[row] = s1; fd[row] = s2; }
}

// ---------------- fused attention: softmax -> WMMA att@Wh -> +res -> LN -> relu ----------------
#define ATT_LD 264   // 256 + 8 pad (bank-conflict avoidance)
__global__ __launch_bounds__(256)
void k_attn(const float* __restrict__ adj,
            const float* __restrict__ fs, const float* __restrict__ fd,
            const bf16_t* __restrict__ whT,
            const float* __restrict__ res,
            const float* __restrict__ lng, const float* __restrict__ lnb,
            float* __restrict__ outf,
            bf16_t* __restrict__ outbf, int bfld, int bfoff) {
  __shared__ bf16_t att_sh[16 * ATT_LD];
  __shared__ float  x_sh[16 * NGP];
  const int lane = threadIdx.x & 31, wave = threadIdx.x >> 5;
  const int half = lane >> 4, l15 = lane & 15;
  const int b = blockIdx.x >> 4, rblk = blockIdx.x & 15;

  // Phase 1: masked-softmax attention rows (2 rows per wave)
  for (int rr = wave * 2; rr < wave * 2 + 2; ++rr) {
    const int rowN = rblk * 16 + rr;
    const size_t m = (size_t)b * NN + rowN;
    const float fsv = fs[m];
    const float* arow = adj + m * NN;
    const float* fdb = fd + (size_t)b * NN;
    float ev[8]; float mx = -3.0e38f;
#pragma unroll
    for (int j = 0; j < 8; ++j) {
      int c = lane + 32 * j;
      float e = fsv + fdb[c];
      e = (e > 0.f) ? e : 0.2f * e;           // LeakyReLU(0.2)
      float a = arow[c];
      float v = (a > 0.f) ? e : NEGV;
      ev[j] = v; mx = fmaxf(mx, v);
    }
    mx = wave_max(mx);
    float sum = 0.f;
#pragma unroll
    for (int j = 0; j < 8; ++j) { ev[j] = __expf(ev[j] - mx); sum += ev[j]; }
    sum = wave_sum(sum);
    float inv = 1.0f / sum;
#pragma unroll
    for (int j = 0; j < 8; ++j)
      att_sh[rr * ATT_LD + lane + 32 * j] = (bf16_t)(ev[j] * inv);
  }
  __syncthreads();

  // Phase 2: h_prime = att(16x256) @ Wh(256xNGP) via WMMA, +residual
  // pass A: each wave computes tiles {wave, wave+8}; shared LDS A-frag, pipelined global B
  const bf16_t* arow = &att_sh[l15 * ATT_LD];
  {
    const int gA = wave * 16 + l15;
    const int gB = (wave + 8) * 16 + l15;
    const bf16_t* browA = whT + ((size_t)b * NGP + gA) * NN;
    const bf16_t* browB = whT + ((size_t)b * NGP + gB) * NN;
    v8f accA = {}, accB = {};
    v16bf bA = load_b_frag(browA, 0, half);
    v16bf bB = load_b_frag(browB, 0, half);
    constexpr int KS = NN / 32;
#pragma unroll
    for (int ks = 0; ks < KS; ++ks) {
      const int k0n = (ks + 1 < KS) ? (ks + 1) * 32 : 0;
      v16bf bAn = load_b_frag(browA, k0n, half);
      v16bf bBn = load_b_frag(browB, k0n, half);
      v16bf a = load_a_frag(arow, ks * 32, half);   // LDS
      accA = wmma_bf16(a, bA, accA);
      accB = wmma_bf16(a, bB, accB);
      bA = bAn; bB = bBn;
    }
#pragma unroll
    for (int r = 0; r < 8; ++r) {
      int ml = r + half * 8;
      size_t m = (size_t)b * NN + rblk * 16 + ml;
      x_sh[ml * NGP + gA] = accA[r] + res[m * NGP + gA];
      x_sh[ml * NGP + gB] = accB[r] + res[m * NGP + gB];
    }
  }
  // pass B: waves 0..3 cover tiles 16..19
  if (wave < 4) {
    const int g = (16 + wave) * 16 + l15;
    const bf16_t* brow = whT + ((size_t)b * NGP + g) * NN;
    v8f acc = {};
    v16bf bb = load_b_frag(brow, 0, half);
    constexpr int KS = NN / 32;
#pragma unroll
    for (int ks = 0; ks < KS; ++ks) {
      const int k0n = (ks + 1 < KS) ? (ks + 1) * 32 : 0;
      v16bf bbn = load_b_frag(brow, k0n, half);
      v16bf a = load_a_frag(arow, ks * 32, half);
      acc = wmma_bf16(a, bb, acc);
      bb = bbn;
    }
#pragma unroll
    for (int r = 0; r < 8; ++r) {
      int ml = r + half * 8;
      size_t m = (size_t)b * NN + rblk * 16 + ml;
      x_sh[ml * NGP + g] = acc[r] + res[m * NGP + g];
    }
  }
  __syncthreads();

  // Phase 3: LayerNorm (over true G=300) + ReLU, write f32 + bf16 (padded cols -> 0)
  for (int rr = wave * 2; rr < wave * 2 + 2; ++rr) {
    size_t m = (size_t)b * NN + rblk * 16 + rr;
    const float* xr = &x_sh[rr * NGP];
    float s = 0.f;
    for (int g = lane; g < NG; g += 32) s += xr[g];
    s = wave_sum(s);
    float mu = s * (1.0f / NG);
    float s2 = 0.f;
    for (int g = lane; g < NG; g += 32) { float d = xr[g] - mu; s2 += d * d; }
    s2 = wave_sum(s2);
    float rstd = rsqrtf(s2 * (1.0f / NG) + LNEPS);
    for (int g = lane; g < NGP; g += 32) {
      float y = 0.f;
      if (g < NG) {
        y = (xr[g] - mu) * rstd * lng[g] + lnb[g];
        y = fmaxf(y, 0.f);
      }
      outf[m * NGP + g] = y;
      outbf[m * (size_t)bfld + bfoff + g] = (bf16_t)y;
    }
  }
}

// ---------------- host side ----------------
extern "C" void kernel_launch(void* const* d_in, const int* in_sizes, int n_in,
                              void* d_out, int out_size, void* d_ws, size_t ws_size,
                              hipStream_t stream) {
  (void)in_sizes; (void)n_in; (void)out_size; (void)ws_size;
  const float* h        = (const float*)d_in[0];
  const float* syn_adj  = (const float*)d_in[1];
  const int*   positions= (const int*)d_in[3];

  // per-layer parameter indices (dict insertion order)
  // syn0: 4..11 (W,asrc,adst,pos,lng,lnb,tW,tb)  syn1: 12..17 (W,asrc,adst,pos,lng,lnb)
  // sem0: 18..25                                  sem1: 26..31
  // fus: 32 (W), 33 (b)
  auto F = [&](int i) { return (const float*)d_in[i]; };

  // workspace bump allocator
  char* wsb = (char*)d_ws; size_t off = 0;
  auto alloc = [&](size_t bytes) -> void* {
    void* p = wsb + off; off = (off + bytes + 255) & ~(size_t)255; return p;
  };
  bf16_t* hbf   = (bf16_t*)alloc((size_t)NM * NH * 2);
  bf16_t* hnbf  = (bf16_t*)alloc((size_t)NM * NH * 2);
  float*  sim   = (float*) alloc((size_t)NB * NN * NN * 4);
  float*  semadj= (float*) alloc((size_t)NB * NN * NN * 4);
  float*  Wh    = (float*) alloc((size_t)NM * NGP * 4);
  bf16_t* WhT   = (bf16_t*)alloc((size_t)NB * NGP * NN * 2);
  float*  resb  = (float*) alloc((size_t)NM * NGP * 4);
  float*  fs    = (float*) alloc((size_t)NM * 4);
  float*  fd    = (float*) alloc((size_t)NM * 4);
  float*  outA  = (float*) alloc((size_t)NM * NGP * 4);   // layer out f32 (residual for next)
  float*  outB  = (float*) alloc((size_t)NM * NGP * 4);   // last-layer f32 (scratch)
  bf16_t* xbf   = (bf16_t*)alloc((size_t)NM * NGP * 2);   // layer out bf16 (next GEMM A)
  bf16_t* fused = (bf16_t*)alloc((size_t)NM * 640 * 2);   // concat bf16 [syn | sem]
  bf16_t* syn0Wt = (bf16_t*)alloc((size_t)NGP * NH * 2);
  bf16_t* syn0tWt= (bf16_t*)alloc((size_t)NGP * NH * 2);
  bf16_t* syn1Wt = (bf16_t*)alloc((size_t)NGP * NGP * 2);
  bf16_t* sem0Wt = (bf16_t*)alloc((size_t)NGP * NH * 2);
  bf16_t* sem0tWt= (bf16_t*)alloc((size_t)NGP * NH * 2);
  bf16_t* sem1Wt = (bf16_t*)alloc((size_t)NGP * NGP * 2);
  bf16_t* fusWt  = (bf16_t*)alloc((size_t)NGP * 640 * 2);

  // ---- prep ----
  k_prep_h<<<NM, 256, 0, stream>>>(h, hbf, hnbf);
  {
    size_t n = (size_t)NB * NN * NN;
    k_zero<<<(int)((n + 255) / 256), 256, 0, stream>>>(semadj, n);
  }
  // ---- semantic graph ----
  k_sim<<<dim3(8, NB), 256, 0, stream>>>(hnbf, sim);
  k_topk<<<NM / 8, 256, 0, stream>>>(sim, semadj);
  k_rownorm<<<NM / 8, 256, 0, stream>>>(semadj);

  // ---- weight prep ----
  auto tpad = [&](const float* W, bf16_t* Wt, int Ksrc, int Kp) {
    size_t n = (size_t)NGP * Kp;
    k_tpad<<<(int)((n + 255) / 256), 256, 0, stream>>>(W, Wt, Ksrc, NG, Kp);
  };
  tpad(F(4),  syn0Wt,  NH, NH);
  tpad(F(10), syn0tWt, NH, NH);
  tpad(F(12), syn1Wt,  NG, NGP);
  tpad(F(18), sem0Wt,  NH, NH);
  tpad(F(24), sem0tWt, NH, NH);
  tpad(F(26), sem1Wt,  NG, NGP);
  {
    size_t n = (size_t)NGP * 640;
    k_tpad_fus<<<(int)((n + 255) / 256), 256, 0, stream>>>(F(32), fusWt);
  }

  const dim3 ggrid(NM / 16 / 8, NGP / 64);    // (256, 5): each wave 16x64

  // ---- one GAT layer helper ----
  auto gat_layer = [&](const float* adjp,
                       const bf16_t* Abf, int lda, int ksteps,
                       const bf16_t* Wt,
                       const float* pos_tab, const float* asrc, const float* adst,
                       const float* lng, const float* lnb,
                       const float* residual,   // f32 [NM,NGP]
                       float* outf, bf16_t* outbf, int bfld, int bfoff) {
    k_gemm<<<ggrid, 256, 0, stream>>>(Abf, lda, Wt, lda == NH ? NH : NGP, Wh, NGP, NGP,
                                      ksteps, positions, pos_tab, nullptr, 0, WhT);
    k_fsfd<<<NM / 8, 256, 0, stream>>>(Wh, asrc, adst, fs, fd);
    k_attn<<<NB * 16, 256, 0, stream>>>(adjp, fs, fd, WhT, residual, lng, lnb,
                                        outf, outbf, bfld, bfoff);
  };

  // ======== syntactic stream ========
  // layer 0: residual = h @ tW + tb
  k_gemm<<<ggrid, 256, 0, stream>>>(hbf, NH, syn0tWt, NH, resb, NGP, NGP,
                                    NH / 32, nullptr, nullptr, F(11), 0, nullptr);
  gat_layer(syn_adj, hbf, NH, NH / 32, syn0Wt, F(7), F(5), F(6), F(8), F(9),
            resb, outA, xbf, NGP, 0);
  // layer 1: residual = layer-0 output
  gat_layer(syn_adj, xbf, NGP, NGP / 32, syn1Wt, F(15), F(13), F(14), F(16), F(17),
            outA, outB, fused, 640, 0);

  // ======== semantic stream ========
  k_gemm<<<ggrid, 256, 0, stream>>>(hbf, NH, sem0tWt, NH, resb, NGP, NGP,
                                    NH / 32, nullptr, nullptr, F(25), 0, nullptr);
  gat_layer(semadj, hbf, NH, NH / 32, sem0Wt, F(21), F(19), F(20), F(22), F(23),
            resb, outA, xbf, NGP, 0);
  gat_layer(semadj, xbf, NGP, NGP / 32, sem1Wt, F(29), F(27), F(28), F(30), F(31),
            outA, outB, fused, 640, 320);

  // ======== fusion: relu(concat @ fus_W + fus_b) -> d_out [NM, 300] ========
  k_gemm<<<ggrid, 256, 0, stream>>>(fused, 640, fusWt, 640, (float*)d_out, NG, NG,
                                    640 / 32, nullptr, nullptr, F(33), 1, nullptr);
}
